// binary_conv_72816875536780
// MI455X (gfx1250) — compile-verified, matrix operand order checked by reading, not run on hardware
//
#include <hip/hip_runtime.h>

typedef __attribute__((ext_vector_type(16))) __bf16 v16bf;
typedef __attribute__((ext_vector_type(8)))  float  v8f;

#define WINDOW 1152          // 3*3*128
#define NF     256
#define NPIX   8192          // 8*32*32
#define KTOT   2304          // 2 residual levels * WINDOW
#define KC     32            // WMMA K step (bf16 16x16x32)
#define NCHUNK (KTOT / KC)   // 72
#define LVLCH  (WINDOW / KC) // 36
#define DPLANE ((size_t)WINDOW * NF)   // 294912 elems per plane

union FragAB { v16bf v; uint4 q[2]; };

__device__ __forceinline__ unsigned short f2bf(float f) {
  unsigned u = __float_as_uint(f);
  u += 0x7FFFu + ((u >> 16) & 1u);      // round-to-nearest-even
  return (unsigned short)(u >> 16);
}
__device__ __forceinline__ unsigned pk2bf(float lo, float hi) {
  return (unsigned)f2bf(lo) | ((unsigned)f2bf(hi) << 16);
}
__device__ __forceinline__ float sgnf(float v) {
  return (v > 0.f) ? 1.f : ((v < 0.f) ? -1.f : 0.f);
}
__device__ __forceinline__ float fetchx(const float* __restrict__ x,
                                        int b, int hh, int ww, int c) {
  bool in = (hh >= 0) & (hh < 32) & (ww >= 0) & (ww < 32);
  int idx = (((b * 32 + hh) * 32 + ww) << 7) + c;
  return in ? x[idx] : 0.f;
}

// ------------------------------------------------------------------
// Kernel 1: binarize+scale+prune coefficients and Walsh-Hadamard them
// over the 8 sign-selector LUT entries of each (level, part) bank.
// Output: D[plane=32][n=256][k=1152] bf16   (k contiguous for B frags)
// plane = level*16 + part*8 + T
// D_T = 1/8 * sum_j (prod_{m in T} sigma_jm) Cb[j]  -- exact in bf16
// ------------------------------------------------------------------
__global__ __launch_bounds__(256) void wht_coeff_kernel(
    const float* __restrict__ c, const float* __restrict__ gamma,
    const float* __restrict__ mask, unsigned short* __restrict__ D) {
  int gid = blockIdx.x * 256 + threadIdx.x;  // 1152*256 threads total
  int k = gid >> 8;
  int n = gid & 255;
  float g  = fabsf(gamma[0]);
  float mk = mask[k * NF + n];
  float cb[32];
#pragma unroll
  for (int j = 0; j < 32; ++j) {
    float v = c[(size_t)j * DPLANE + (size_t)k * NF + n];
    cb[j] = sgnf(v) * g * mk;               // binarize * |gamma| * prune
  }
#pragma unroll
  for (int p = 0; p < 4; ++p) {             // p = level*2 + part
#pragma unroll
    for (int T = 0; T < 8; ++T) {
      float s = 0.f;
#pragma unroll
      for (int j = 0; j < 8; ++j) {
        // idx bit2<->s0, bit1<->s1, bit0<->s2 ; T bit0<->s0, bit1<->s1, bit2<->s2
        int jm = ((j >> 2) & 1) | (((j >> 1) & 1) << 1) | ((j & 1) << 2);
        s += (__popc(jm & T) & 1) ? -cb[p * 8 + j] : cb[p * 8 + j];
      }
      int plane = p * 8 + T;
      D[(size_t)plane * DPLANE + (size_t)n * WINDOW + k] = f2bf(0.125f * s);
    }
  }
}

// ------------------------------------------------------------------
// Kernel 2: fused dense-LUT-plane construction + bf16 WMMA GEMM.
// Block tile: 64 pixels x 256 outputs. 8 waves = 2 M-groups x 4 N-quarters;
// each wave: 2 M-subtiles x 4 N-tiles -> every B fragment feeds 2 WMMAs
// (1 global_load_b128 per WMMA), halving the D traffic vs a 1x8 mapping.
// ------------------------------------------------------------------
__global__ __launch_bounds__(256) void lut_gemm_kernel(
    const float* __restrict__ x, const unsigned short* __restrict__ D,
    const int* __restrict__ m0, const int* __restrict__ m1,
    const int* __restrict__ m2, float* __restrict__ out) {
  __shared__ unsigned int A32[16 * 64 * (KC / 2)];  // 64 KB, bf16 pairs packed

  const int tid  = threadIdx.x;
  const int lane = tid & 31;
  const int wv   = tid >> 5;
  const int mg   = wv & 1;       // M group: pixel rows mg*32 .. mg*32+31
  const int nq   = wv >> 1;      // N quarter: cols nq*64 .. nq*64+63
  const int hi   = lane >> 4;
  const int l15  = lane & 15;
  const int pix0 = blockIdx.x * 64;

  v8f acc[8];                    // [mi*4 + nt]
  {
    v8f z = {0.f, 0.f, 0.f, 0.f, 0.f, 0.f, 0.f, 0.f};
#pragma unroll
    for (int t = 0; t < 8; ++t) acc[t] = z;
  }

  const int kp   = (tid & 15) * 2;  // even k-pair owned by this thread
  const int prow = tid >> 4;        // covers pixels prow + 16*i, i=0..3

  for (int ch = 0; ch < NCHUNK; ++ch) {
    const int lvl  = (ch >= LVLCH) ? 1 : 0;
    const int kloc = ch * KC - lvl * WINDOW;

    // ---- build the 16 dense LUT-plane A tiles in LDS (packed b32 stores) ----
    {
      int ci[2], kwi[2], khi[2];
      int gc[2][3], gw[2][3], gh[2][3];
#pragma unroll
      for (int u = 0; u < 2; ++u) {
        const int kw = kloc + kp + u;
        ci[u] = kw & 127;
        const int tt = kw >> 7;
        kwi[u] = tt % 3; khi[u] = tt / 3;
        const int g[3] = {m0[kw], m1[kw], m2[kw]};
#pragma unroll
        for (int mI = 0; mI < 3; ++mI) {
          gc[u][mI] = g[mI] & 127;
          const int uu = g[mI] >> 7;
          gw[u][mI] = uu % 3; gh[u][mI] = uu / 3;
        }
      }
#pragma unroll
      for (int i = 0; i < 4; ++i) {
        const int pl  = prow + i * 16;
        const int pix = pix0 + pl;
        const int b = pix >> 10, h = (pix >> 5) & 31, w = pix & 31;
        float rp[2], rn[2], mu[2][8];
#pragma unroll
        for (int u = 0; u < 2; ++u) {
          float v  = fetchx(x, b, h + khi[u] - 1,   w + kwi[u] - 1,   ci[u]);
          float q0 = fetchx(x, b, h + gh[u][0] - 1, w + gw[u][0] - 1, gc[u][0]);
          float q1 = fetchx(x, b, h + gh[u][1] - 1, w + gw[u][1] - 1, gc[u][1]);
          float q2 = fetchx(x, b, h + gh[u][2] - 1, w + gw[u][2] - 1, gc[u][2]);
          if (lvl) {               // residual level 2: r = x - sign(x)
            v  -= sgnf(v);  q0 -= sgnf(q0);
            q1 -= sgnf(q1); q2 -= sgnf(q2);
          }
          const float s0 = sgnf(q0), s1 = sgnf(q1), s2 = sgnf(q2);
          rp[u] = fmaxf(v, 0.f);  rn[u] = fmaxf(-v, 0.f);
          mu[u][0] = 1.f;      mu[u][1] = s0;
          mu[u][2] = s1;       mu[u][3] = s0 * s1;
          mu[u][4] = s2;       mu[u][5] = s0 * s2;
          mu[u][6] = s1 * s2;  mu[u][7] = mu[u][3] * s2;
        }
#pragma unroll
        for (int T = 0; T < 8; ++T) {
          A32[((T)     * 64 + pl) * (KC / 2) + (kp >> 1)] =
              pk2bf(rp[0] * mu[0][T], rp[1] * mu[1][T]);
          A32[((T + 8) * 64 + pl) * (KC / 2) + (kp >> 1)] =
              pk2bf(rn[0] * mu[0][T], rn[1] * mu[1][T]);
        }
      }
    }
    __syncthreads();

    // prefetch next chunk's D rows into the cache hierarchy
    if (ch + 1 < NCHUNK) {
      const int nl = ((ch + 1) >= LVLCH) ? 1 : 0;
      const unsigned short* pf = D + (size_t)(nl * 16) * DPLANE
          + (size_t)(nq * 64 + l15) * WINDOW + ((ch + 1) * KC - nl * WINDOW);
      __builtin_prefetch(pf, 0, 1);
    }

    // ---- 16-plane WMMA accumulate: 2 A frags x 4 B frags -> 8 WMMAs ----
    const int row0 = mg * 32 + l15;       // first M-subtile row
    for (int qp = 0; qp < 16; ++qp) {
      FragAB a0, a1;
      const uint4* ap0 =
          reinterpret_cast<const uint4*>(&A32[(qp * 64 + row0) * (KC / 2)]);
      const uint4* ap1 =
          reinterpret_cast<const uint4*>(&A32[(qp * 64 + row0 + 16) * (KC / 2)]);
      a0.q[0] = ap0[hi];  a0.q[1] = ap0[hi + 2];   // K0..7/K8..15 , +16
      a1.q[0] = ap1[hi];  a1.q[1] = ap1[hi + 2];
      const int plane = lvl * 16 + qp;
      const unsigned short* drow = D + (size_t)plane * DPLANE
          + (size_t)(nq * 64 + l15) * WINDOW + kloc + hi * 8;
#pragma unroll
      for (int nt = 0; nt < 4; ++nt) {
        FragAB bm;
        const uint4* bp =
            reinterpret_cast<const uint4*>(drow + (size_t)nt * 16 * WINDOW);
        bm.q[0] = bp[0];
        bm.q[1] = bp[2];
        acc[nt] = __builtin_amdgcn_wmma_f32_16x16x32_bf16(
            false, a0.v, false, bm.v, (short)0, acc[nt], false, false);
        acc[4 + nt] = __builtin_amdgcn_wmma_f32_16x16x32_bf16(
            false, a1.v, false, bm.v, (short)0, acc[4 + nt], false, false);
      }
    }
    __syncthreads();
  }

  // ---- epilogue: C/D layout = VGPR i -> M = i + 8*hi, N = lane&15 ----
#pragma unroll
  for (int mi = 0; mi < 2; ++mi) {
#pragma unroll
    for (int nt = 0; nt < 4; ++nt) {
      const int col = nq * 64 + nt * 16 + l15;
#pragma unroll
      for (int i = 0; i < 8; ++i) {
        const int row = pix0 + mg * 32 + mi * 16 + hi * 8 + i;
        out[(size_t)row * NF + col] = acc[mi * 4 + nt][i];
      }
    }
  }
}

extern "C" void kernel_launch(void* const* d_in, const int* in_sizes, int n_in,
                              void* d_out, int out_size, void* d_ws, size_t ws_size,
                              hipStream_t stream) {
  const float* x     = (const float*)d_in[0];
  const float* c     = (const float*)d_in[1];
  const float* gamma = (const float*)d_in[2];
  const float* mask  = (const float*)d_in[3];
  const int*   r0    = (const int*)d_in[4];
  const int*   r1    = (const int*)d_in[5];
  const int*   r2    = (const int*)d_in[6];
  (void)in_sizes; (void)n_in; (void)out_size; (void)ws_size;

  unsigned short* D = (unsigned short*)d_ws;   // 32*1152*256*2 B = 18.9 MB

  wht_coeff_kernel<<<WINDOW, 256, 0, stream>>>(c, gamma, mask, D);
  lut_gemm_kernel<<<NPIX / 64, 256, 0, stream>>>(x, D, r0, r1, r2, (float*)d_out);
}
